// Attention_34308198760808
// MI455X (gfx1250) — matrix operand order repaired and checked by reading, not compiled, not run on hardware
//
#include <hip/hip_runtime.h>
#include <hip/hip_bf16.h>

typedef __attribute__((ext_vector_type(16))) _Float16 v16h;
typedef __attribute__((ext_vector_type(8)))  float    v8f;

#define DIMC 96
#define NHEAD 3
#define HDIM 32
#define WSZ 8
#define NTOK 64
#define RESO 256
#define SSH 4

// ---------------- workspace layout (bytes) ----------------
#define WQKV_OFF   0ull                     // 288*96 f16   = 55296
#define QKVB_OFF   55296ull                 // 288 f32      = 1152
#define CCF_OFF    56576ull                 // 96*96 f16    = 18432
#define PROJW_OFF  75264ull                 // 96*96 f16    = 18432
#define BIAS_OFF   93696ull                 // 3*64*64 f32  = 49152
#define VG_OFF     143360ull                // 8*96*256*256 f32 = 201326592
#define XO_OFF     (VG_OFF + 201326592ull)  // 8*256*256*96 f32 = 201326592

// ---------------- WMMA fragment loaders ----------------
// A-matrix 16x32 f16 (ISA table): lane<16 -> K {0..7,16..23}, lane>=16 -> K {8..15,24..31}
__device__ __forceinline__ v16h frag_a_f16(const _Float16* base, int ld) {
  int lane = threadIdx.x & 31;
  const _Float16* p = base + (lane & 15) * ld + ((lane & 16) ? 8 : 0);
  v16h a;
#pragma unroll
  for (int e = 0; e < 8; ++e) a[e] = p[e];
#pragma unroll
  for (int e = 0; e < 8; ++e) a[8 + e] = p[16 + e];
  return a;
}
// same mapping, source is f32 (convert on the fly; used for softmaxed P)
__device__ __forceinline__ v16h frag_a_f32(const float* base, int ld) {
  int lane = threadIdx.x & 31;
  const float* p = base + (lane & 15) * ld + ((lane & 16) ? 8 : 0);
  v16h a;
#pragma unroll
  for (int e = 0; e < 8; ++e) a[e] = (_Float16)p[e];
#pragma unroll
  for (int e = 0; e < 8; ++e) a[8 + e] = (_Float16)p[16 + e];
  return a;
}
// B-matrix 32x16 f16: lane = n, lane<16 -> K {0..7,16..23}, lane>=16 -> K {8..15,24..31}
__device__ __forceinline__ v16h frag_b_f16(const _Float16* base, int ld) {
  int lane = threadIdx.x & 31;
  const _Float16* p = base + (lane & 15) * ld + ((lane & 16) ? 8 : 0);
  v16h b;
#pragma unroll
  for (int e = 0; e < 8; ++e) b[e] = p[e];
#pragma unroll
  for (int e = 0; e < 8; ++e) b[8 + e] = p[16 + e];
  return b;
}

__device__ __forceinline__ v8f wmma_f16(v16h a, v16h b, v8f c) {
  return __builtin_amdgcn_wmma_f32_16x16x32_f16(false, a, false, b, (short)0, c,
                                                false, false);
}

// analytic Swin shift mask (cloud-mask aliasing folded in)
__device__ __forceinline__ float mask_val(int n, int m, bool lastRow, bool lastCol) {
  if (!lastRow && !lastCol) return 0.f;
  if (lastRow != lastCol) return -100.f;  // edge windows: fully masked by cloud trick
  int ln = ((n >> 3) < SSH ? 0 : 1) * 2 + ((n & 7) < SSH ? 0 : 1);
  int lm = ((m >> 3) < SSH ? 0 : 1) * 2 + ((m & 7) < SSH ? 0 : 1);
  return (ln == lm) ? 0.f : -100.f;
}

// ---------------- init: weights -> f16, fold q-scale and cc concat ----------------
__global__ __launch_bounds__(256) void k_init_w(
    const float* __restrict__ Vw, const float* __restrict__ Vb,
    const float* __restrict__ QKw, const float* __restrict__ QKb,
    const float* __restrict__ ccw, const float* __restrict__ pw,
    _Float16* __restrict__ wqkv, float* __restrict__ qkvb,
    _Float16* __restrict__ ccf, _Float16* __restrict__ projw) {
  const float qs = 0.17677669529663687f;  // 32^-0.5 folded into q rows
  int tid = threadIdx.x;
  for (int e = tid; e < 288 * DIMC; e += 256) {
    int o = e / DIMC, c = e - o * DIMC;
    float w;
    if (o < 96)        w = QKw[o * DIMC + c] * qs;
    else if (o < 192)  w = QKw[o * DIMC + c];
    else               w = Vw[(o - 192) * DIMC + c];
    wqkv[e] = (_Float16)w;
  }
  for (int e = tid; e < 288; e += 256)
    qkvb[e] = (e < 96) ? QKb[e] * qs : (e < 192 ? QKb[e] : Vb[e - 192]);
  for (int e = tid; e < DIMC * DIMC; e += 256) {
    int o = e / DIMC, c = e - o * DIMC;
    ccf[e]   = (_Float16)(ccw[o * 2 * DIMC + c] + ccw[o * 2 * DIMC + DIMC + c]);
    projw[e] = (_Float16)pw[e];
  }
}

// ---------------- init: relative-position-bias MLP table [3][64][64] ----------------
__global__ __launch_bounds__(256) void k_init_bias(
    const float* __restrict__ mw1, const float* __restrict__ mb1,
    const float* __restrict__ mw2, const float* __restrict__ mb2,
    float* __restrict__ biasTab) {
  int pr = blockIdx.x * 256 + threadIdx.x;  // 16 blocks x 256 = 4096 pairs
  if (pr < NTOK * NTOK) {
    int n = pr >> 6, m = pr & 63;
    float dy = (float)((n >> 3) - (m >> 3));
    float dx = (float)((n & 7) - (m & 7));
    float ry = (dy > 0.f ? 1.f : (dy < 0.f ? -1.f : 0.f)) * log1pf(fabsf(dy));
    float rx = (dx > 0.f ? 1.f : (dx < 0.f ? -1.f : 0.f)) * log1pf(fabsf(dx));
    float a0 = mb2[0], a1 = mb2[1], a2 = mb2[2];
    for (int j = 0; j < 256; ++j) {
      float hv = fmaxf(ry * mw1[j * 2] + rx * mw1[j * 2 + 1] + mb1[j], 0.f);
      a0 += hv * mw2[j];
      a1 += hv * mw2[256 + j];
      a2 += hv * mw2[512 + j];
    }
    biasTab[pr] = a0;
    biasTab[4096 + pr] = a1;
    biasTab[8192 + pr] = a2;
  }
}

// ---------------- kernel 1: shift + QKV + window attention + cc, per window ----------------
__global__ __launch_bounds__(256) void k_attn(
    const float* __restrict__ x, const _Float16* __restrict__ wqkv,
    const float* __restrict__ qkvb, const _Float16* __restrict__ ccf,
    const float* __restrict__ ccb, const float* __restrict__ biasTab,
    float* __restrict__ Vg, float* __restrict__ xo) {
  // per-head: q[64][32] | k[64][32] | v[32][64] (f16), heads packed; attn scratch f32
  __shared__ _Float16 sm_qkv[3 * 6144];      // 36864 B (also f32 staging at phase 0)
  __shared__ float    sm_attn[NTOK * NTOK];  // 16384 B (also reused as xT f16)
  _Float16* sm_xT = (_Float16*)sm_attn;      // [64][96] token-major (12288 B)
  float* sm_stage = (float*)sm_qkv;          // [96][64] f32 staging (24576 B)

  const int tid = threadIdx.x;
  const int lane = tid & 31;
  const int wave = tid >> 5;
  const int hiOff = (lane & 16) ? 8 : 0;

  const int win = blockIdx.x;
  const int b = win >> 10;
  const int wh = (win & 1023) >> 5;
  const int ww = win & 31;
  const int i0 = wh * WSZ, j0 = ww * WSZ;
  const bool lastRow = (wh == 31), lastCol = (ww == 31);

  __builtin_prefetch(wqkv, 0, 3);
  __builtin_prefetch(ccf, 0, 3);

  // ---- async gather of shifted window pixels into LDS staging (fp32) ----
  {
    unsigned lds_base = (unsigned)(uintptr_t)(void*)sm_stage;
    const char* xb = (const char*)x;
#pragma unroll
    for (int i = 0; i < 12; ++i) {
      int e = tid + i * 256;  // [96 ch][64 tok] c-major
      int c = e >> 6, t = e & 63;
      int ii = (i0 + (t >> 3) + SSH) & 255;
      int jj = (j0 + (t & 7) + SSH) & 255;
      unsigned goff = ((((unsigned)b * DIMC + c) * RESO + ii) * RESO + jj) * 4u;
      unsigned laddr = lds_base + (unsigned)e * 4u;
      asm volatile("global_load_async_to_lds_b32 %0, %1, %2"
                   :: "v"(laddr), "v"(goff), "s"(xb) : "memory");
    }
    asm volatile("s_wait_asynccnt 0" ::: "memory");
  }
  __syncthreads();
  // ---- convert staging f32 -> token-major f16 xT ----
  for (int e = tid; e < DIMC * NTOK; e += 256) {
    int c = e >> 6, t = e & 63;
    sm_xT[t * DIMC + c] = (_Float16)sm_stage[e];
  }
  __syncthreads();

  // ---- QKV GEMM: [288,96] x [96,64], 72 tiles (9/wave), K=96 ----
  for (int it = 0; it < 9; ++it) {
    int tt = wave + it * 8;
    int m0 = (tt >> 2) << 4, n0 = (tt & 3) << 4;
    v8f acc = {};
#pragma unroll
    for (int kc = 0; kc < 3; ++kc) {
      v16h a = frag_a_f16(wqkv + m0 * DIMC + kc * 32, DIMC);
      v16h bb = frag_b_f16(sm_xT + n0 * DIMC + kc * 32, DIMC);
      acc = wmma_f16(a, bb, acc);
    }
    int t = n0 + (lane & 15);
    int o0 = m0 + hiOff;          // lane's first output row
    int sec = m0 / 96;            // tile-uniform: 0=q,1=k,2=v (16 | 96)
    int oo0 = o0 - sec * 96;      // multiple of 8 -> head/d-base uniform for the span
    int h = oo0 >> 5, d0 = oo0 & 31;
    const float4 qb0 = *(const float4*)(qkvb + o0);
    const float4 qb1 = *(const float4*)(qkvb + o0 + 4);
    float qb[8] = {qb0.x, qb0.y, qb0.z, qb0.w, qb1.x, qb1.y, qb1.z, qb1.w};
    if (sec < 2) {                // q / k: token-major [64][32]
      _Float16* dst = sm_qkv + h * 6144 + sec * 2048 + t * 32 + d0;
#pragma unroll
      for (int r = 0; r < 8; ++r) dst[r] = (_Float16)(acc[r] + qb[r]);
    } else {                      // v: d-major [32][64] + global V (shifted frame)
      _Float16* dstv = sm_qkv + h * 6144 + 4096 + d0 * 64 + t;
      float* vg = Vg + (((size_t)b * DIMC + oo0) * RESO + i0 + (t >> 3)) * RESO +
                  j0 + (t & 7);
#pragma unroll
      for (int r = 0; r < 8; ++r) {
        float val = acc[r] + qb[r];
        dstv[r * 64] = (_Float16)val;
        vg[(size_t)r * RESO * RESO] = val;
      }
    }
  }
  __syncthreads();

  // ---- per-head attention ----
  for (int h = 0; h < NHEAD; ++h) {
    const _Float16* qh = sm_qkv + h * 6144;
    const _Float16* kh = qh + 2048;
    const _Float16* vh = qh + 4096;

    // Q K^T : 16 tiles (2/wave), K=32
    for (int it = 0; it < 2; ++it) {
      int tt = wave * 2 + it;
      int m0 = (tt >> 2) << 4, n0 = (tt & 3) << 4;
      v16h a = frag_a_f16(qh + m0 * 32, 32);
      v16h bb = frag_b_f16(kh + n0 * 32, 32);
      v8f acc = {};
      acc = wmma_f16(a, bb, acc);
      int mcol = n0 + (lane & 15);
#pragma unroll
      for (int r = 0; r < 8; ++r) {
        int nrow = m0 + hiOff + r;
        float val = acc[r] + biasTab[h * 4096 + nrow * 64 + mcol] +
                    mask_val(nrow, mcol, lastRow, lastCol);
        sm_attn[nrow * 64 + mcol] = val;
      }
    }
    __syncthreads();

    // softmax rows (in place, f32)
    if (tid < NTOK) {
      float mx = -1e30f;
      for (int m2 = 0; m2 < 64; ++m2) mx = fmaxf(mx, sm_attn[tid * 64 + m2]);
      float s = 0.f;
      for (int m2 = 0; m2 < 64; ++m2) {
        float ev = __expf(sm_attn[tid * 64 + m2] - mx);
        sm_attn[tid * 64 + m2] = ev;
        s += ev;
      }
      float inv = 1.f / s;
      for (int m2 = 0; m2 < 64; ++m2) sm_attn[tid * 64 + m2] *= inv;
    }
    __syncthreads();

    // P @ V : out [64 tok][32 d], 8 tiles (1/wave), K=64; write over dead q region
    {
      int m0 = (wave >> 1) << 4, n0 = (wave & 1) << 4;
      v8f acc = {};
#pragma unroll
      for (int kc = 0; kc < 2; ++kc) {
        v16h a = frag_a_f32(sm_attn + m0 * 64 + kc * 32, 64);
        v16h bb = frag_b_f16(vh + n0 * 64 + kc * 32, 64);
        acc = wmma_f16(a, bb, acc);
      }
      _Float16* xatt = (_Float16*)qh;  // [64][32] per head
      int dcol = n0 + (lane & 15);
#pragma unroll
      for (int r = 0; r < 8; ++r)
        xatt[(m0 + hiOff + r) * 32 + dcol] = (_Float16)acc[r];
    }
    __syncthreads();
  }

  // ---- cc 1x1 (concat folded): [96,96] x [96,64], 24 tiles (3/wave) ----
  for (int it = 0; it < 3; ++it) {
    int tt = wave + it * 8;
    int m0 = (tt >> 2) << 4, n0 = (tt & 3) << 4;
    v8f acc = {};
#pragma unroll
    for (int kc = 0; kc < 3; ++kc) {
      v16h a = frag_a_f16(ccf + m0 * DIMC + kc * 32, DIMC);
      v16h bb = frag_b_f16(sm_qkv + kc * 6144 + n0 * 32, 32);  // xatt head kc
      acc = wmma_f16(a, bb, acc);
    }
    int t = n0 + (lane & 15);
    int ii = (i0 + (t >> 3) + SSH) & 255;
    int jj = (j0 + (t & 7) + SSH) & 255;
    float* dst = xo + (((size_t)b * RESO + ii) * RESO + jj) * DIMC;
    int o0 = m0 + hiOff;  // 8 contiguous channels, 32B aligned
    float4 r0, r1;
    r0.x = acc[0] + ccb[o0 + 0];
    r0.y = acc[1] + ccb[o0 + 1];
    r0.z = acc[2] + ccb[o0 + 2];
    r0.w = acc[3] + ccb[o0 + 3];
    r1.x = acc[4] + ccb[o0 + 4];
    r1.y = acc[5] + ccb[o0 + 5];
    r1.z = acc[6] + ccb[o0 + 6];
    r1.w = acc[7] + ccb[o0 + 7];
    *(float4*)(dst + o0) = r0;
    *(float4*)(dst + o0 + 4) = r1;
  }
}

// ---------------- kernel 2: depthwise 5x5 (reflect) + xo add + proj ----------------
__global__ __launch_bounds__(256) void k_conv(
    const float* __restrict__ Vg, const float* __restrict__ xo,
    const float* __restrict__ dww, const float* __restrict__ dwb,
    const _Float16* __restrict__ projw, const float* __restrict__ projb,
    float* __restrict__ out) {
  __shared__ float    sm_patch[32 * 12 * 12];  // 18432 B
  __shared__ float    sm_dw[32 * 25];          // 3200 B
  __shared__ _Float16 sm_y[NTOK * DIMC];       // 12288 B

  const int tid = threadIdx.x;
  const int lane = tid & 31;
  const int wave = tid >> 5;
  const int hiOff = (lane & 16) ? 8 : 0;

  const int blk = blockIdx.x;
  const int b = blk >> 10;
  const int i0 = ((blk & 1023) >> 5) * 8;
  const int j0 = (blk & 31) * 8;

  __builtin_prefetch(projw, 0, 3);

  for (int c0 = 0; c0 < DIMC; c0 += 32) {
    for (int e = tid; e < 32 * 25; e += 256) sm_dw[e] = dww[c0 * 25 + e];
    for (int e = tid; e < 32 * 144; e += 256) {
      int c = e / 144, s = e - c * 144;
      int ai = i0 - 2 + s / 12;
      int aj = j0 - 2 + s % 12;
      ai = ai < 0 ? -ai : (ai > 255 ? 510 - ai : ai);  // reflect pad
      aj = aj < 0 ? -aj : (aj > 255 ? 510 - aj : aj);
      sm_patch[e] = Vg[(((size_t)b * DIMC + c0 + c) * RESO + ai) * RESO + aj];
    }
    __syncthreads();
    for (int e = tid; e < 32 * 64; e += 256) {
      int cl = e >> 6, p = e & 63;
      int py = p >> 3, px = p & 7;
      float s = dwb[c0 + cl];
#pragma unroll
      for (int u = 0; u < 5; ++u)
#pragma unroll
        for (int v = 0; v < 5; ++v)
          s += sm_dw[cl * 25 + u * 5 + v] * sm_patch[cl * 144 + (py + u) * 12 + px + v];
      s += xo[(((size_t)b * RESO + i0 + py) * RESO + j0 + px) * DIMC + c0 + cl];
      sm_y[p * DIMC + c0 + cl] = (_Float16)s;
    }
    __syncthreads();
  }

  // proj: [96,96] x [96,64], 24 tiles (3/wave), K=96
  for (int it = 0; it < 3; ++it) {
    int tt = wave + it * 8;
    int m0 = (tt >> 2) << 4, n0 = (tt & 3) << 4;
    v8f acc = {};
#pragma unroll
    for (int kc = 0; kc < 3; ++kc) {
      v16h a = frag_a_f16(projw + m0 * DIMC + kc * 32, DIMC);
      v16h bb = frag_b_f16(sm_y + n0 * DIMC + kc * 32, DIMC);
      acc = wmma_f16(a, bb, acc);
    }
    int p = n0 + (lane & 15);
    int py = p >> 3, px = p & 7;
#pragma unroll
    for (int r = 0; r < 8; ++r) {
      int o = m0 + hiOff + r;
      out[(((size_t)b * DIMC + o) * RESO + i0 + py) * RESO + j0 + px] =
          acc[r] + projb[o];
    }
  }
}

extern "C" void kernel_launch(void* const* d_in, const int* in_sizes, int n_in,
                              void* d_out, int out_size, void* d_ws, size_t ws_size,
                              hipStream_t stream) {
  const float* x   = (const float*)d_in[0];
  const float* Vw  = (const float*)d_in[1];
  const float* Vb  = (const float*)d_in[2];
  const float* QKw = (const float*)d_in[3];
  const float* QKb = (const float*)d_in[4];
  const float* mw1 = (const float*)d_in[5];
  const float* mb1 = (const float*)d_in[6];
  const float* mw2 = (const float*)d_in[7];
  const float* mb2 = (const float*)d_in[8];
  const float* ccw = (const float*)d_in[9];
  const float* ccb = (const float*)d_in[10];
  const float* dww = (const float*)d_in[11];
  const float* dwb = (const float*)d_in[12];
  const float* pw  = (const float*)d_in[13];
  const float* pb  = (const float*)d_in[14];

  char* ws = (char*)d_ws;
  _Float16* wqkv   = (_Float16*)(ws + WQKV_OFF);
  float*    qkvb   = (float*)   (ws + QKVB_OFF);
  _Float16* ccf    = (_Float16*)(ws + CCF_OFF);
  _Float16* projw  = (_Float16*)(ws + PROJW_OFF);
  float*    biasTab= (float*)   (ws + BIAS_OFF);
  float*    Vg     = (float*)   (ws + VG_OFF);
  float*    xoB    = (float*)   (ws + XO_OFF);
  float*    out    = (float*)d_out;

  k_init_w<<<1, 256, 0, stream>>>(Vw, Vb, QKw, QKb, ccw, pw, wqkv, qkvb, ccf, projw);
  k_init_bias<<<16, 256, 0, stream>>>(mw1, mb1, mw2, mb2, biasTab);
  k_attn<<<8192, 256, 0, stream>>>(x, wqkv, qkvb, ccf, ccb, biasTab, Vg, xoB);
  k_conv<<<8192, 256, 0, stream>>>(Vg, xoB, dww, dwb, projw, pb, out);
}
